// BertNer_10359461118298
// MI455X (gfx1250) — matrix operand order repaired and checked by reading, not compiled, not run on hardware
//
#include <hip/hip_runtime.h>

// ---------------- problem constants (BERT-base, B=16 S=512) ----------------
constexpr int B_  = 16;
constexpr int S_  = 512;
constexpr int H_  = 768;
constexpr int L_  = 12;
constexpr int NH_ = 12;
constexpr int FF_ = 3072;
constexpr int NL_ = 9;
constexpr int DH_ = 64;
constexpr int M_  = B_ * S_;   // 8192 tokens

// ---------------- WMMA fragment types (wave32, gfx1250) ----------------
typedef __attribute__((ext_vector_type(16))) __bf16          v16bf;
typedef __attribute__((ext_vector_type(8)))  float           v8f;
typedef __attribute__((ext_vector_type(8)))  unsigned short  v8us;

union Frag16 { v8us h[2]; v16bf v; };

__device__ inline v8f wmma_bf16(const Frag16& a, const Frag16& b, v8f c) {
  return __builtin_amdgcn_wmma_f32_16x16x32_bf16(false, a.v, false, b.v,
                                                 (short)0, c, false, false);
}

__device__ inline unsigned short f2bf(float f) {
  unsigned int u = __float_as_uint(f);
  unsigned int r = u + 0x7FFFu + ((u >> 16) & 1u);   // round-to-nearest-even
  return (unsigned short)(r >> 16);
}

// ---------------- CDNA5 async global->LDS copy (ASYNCcnt-tracked) ----------------
// the LDS destination address is taken from a VGPR; generic LDS pointers have the byte offset in
// the low 32 bits.
__device__ inline void async_ld_b128(unsigned lds_off, const unsigned short* g) {
  asm volatile("global_load_async_to_lds_b128 %0, %1, off"
               :: "v"(lds_off), "v"(g) : "memory");
}
#define ASYNC_WAIT0() asm volatile("s_wait_asynccnt 0x0" ::: "memory")

// ---------------- block reduction helper (256 threads, 8 waves) ----------------
__device__ inline float block_sum256(float v, float* red8) {
  #pragma unroll
  for (int off = 16; off >= 1; off >>= 1) v += __shfl_xor(v, off, 32);
  const int t = threadIdx.x;
  __syncthreads();
  if ((t & 31) == 0) red8[t >> 5] = v;
  __syncthreads();
  if (t == 0) {
    float s = 0.f;
    #pragma unroll
    for (int i = 0; i < 8; ++i) s += red8[i];
    red8[0] = s;
  }
  __syncthreads();
  return red8[0];
}

// ---------------- embeddings + layernorm ----------------
__global__ __launch_bounds__(256) void embed_ln_kernel(
    const int* __restrict__ ids, const int* __restrict__ tids,
    const float* __restrict__ wemb, const float* __restrict__ pemb,
    const float* __restrict__ temb, const float* __restrict__ g,
    const float* __restrict__ bta, float* __restrict__ hOut,
    unsigned short* __restrict__ xOut)
{
  __shared__ float red8[8];
  const int tok = blockIdx.x;
  const int s   = tok & (S_ - 1);
  const int id  = ids[tok];
  const int tp  = tids[tok];
  const float* w  = wemb + (size_t)id * H_;
  const float* p  = pemb + (size_t)s  * H_;
  const float* tt = temb + (size_t)tp * H_;

  float vals[3]; float sum = 0.f;
  #pragma unroll
  for (int j = 0; j < 3; ++j) {
    int i = threadIdx.x + j * 256;
    vals[j] = w[i] + p[i] + tt[i];
    sum += vals[j];
  }
  float mean = block_sum256(sum, red8) * (1.0f / H_);
  float q = 0.f;
  #pragma unroll
  for (int j = 0; j < 3; ++j) { float d = vals[j] - mean; q += d * d; }
  float rstd = rsqrtf(block_sum256(q, red8) * (1.0f / H_) + 1e-12f);
  #pragma unroll
  for (int j = 0; j < 3; ++j) {
    int i = threadIdx.x + j * 256;
    float y = (vals[j] - mean) * rstd * g[i] + bta[i];
    hOut[(size_t)tok * H_ + i] = y;
    xOut[(size_t)tok * H_ + i] = f2bf(y);
  }
}

// ---------------- layernorm (f32 in -> f32 + bf16 out) ----------------
__global__ __launch_bounds__(256) void ln_kernel(
    const float* __restrict__ in, const float* __restrict__ g,
    const float* __restrict__ bb, float* __restrict__ hOut,
    unsigned short* __restrict__ xOut)
{
  __shared__ float red8[8];
  const int tok = blockIdx.x;
  const float* x = in + (size_t)tok * H_;
  float vals[3]; float s = 0.f;
  #pragma unroll
  for (int j = 0; j < 3; ++j) { vals[j] = x[threadIdx.x + j * 256]; s += vals[j]; }
  float mean = block_sum256(s, red8) * (1.0f / H_);
  float q = 0.f;
  #pragma unroll
  for (int j = 0; j < 3; ++j) { float d = vals[j] - mean; q += d * d; }
  float rstd = rsqrtf(block_sum256(q, red8) * (1.0f / H_) + 1e-12f);
  #pragma unroll
  for (int j = 0; j < 3; ++j) {
    int i = threadIdx.x + j * 256;
    float y = (vals[j] - mean) * rstd * g[i] + bb[i];
    hOut[(size_t)tok * H_ + i] = y;
    xOut[(size_t)tok * H_ + i] = f2bf(y);
  }
}

// ---------------- weight convert + transpose: W[K][N] f32 -> Wt[N][K] bf16 ----------------
__global__ __launch_bounds__(256) void convert_wt_kernel(
    const float* __restrict__ W, unsigned short* __restrict__ Wt, int K, int N)
{
  __shared__ float tile[32][33];
  const int kb = blockIdx.y * 32, nb = blockIdx.x * 32;
  const int tx = threadIdx.x & 31, ty = threadIdx.x >> 5;
  #pragma unroll
  for (int r = ty; r < 32; r += 8)
    tile[r][tx] = W[(size_t)(kb + r) * N + nb + tx];
  __syncthreads();
  #pragma unroll
  for (int r = ty; r < 32; r += 8)
    Wt[(size_t)(nb + r) * K + kb + tx] = f2bf(tile[tx][r]);
}

// ---------------- NT GEMM:  C[M][N] = A[M][K](bf16) * Bt[N][K](bf16)^T ----------------
// 128x64 block tile, 8 waves x (16x64) wave tiles, K-step 32.
// Double-buffered LDS filled with async global->LDS b128 copies (ASYNCcnt).
// EPI: 0 = +bias -> bf16 row-major      (Q, K projections)
//      1 = +bias -> bf16 V-transposed [B][NH][DH][S]   (V projection)
//      2 = +bias + resid -> f32 row-major              (pre-layernorm)
//      3 = +bias, GELU -> bf16 row-major               (FF1)
template <int EPI>
__global__ __launch_bounds__(256) void gemm_nt_bf16(
    const unsigned short* __restrict__ A,
    const unsigned short* __restrict__ Bt,
    const float* __restrict__ bias,
    const float* __restrict__ resid,
    float* __restrict__ outF,
    unsigned short* __restrict__ outB,
    int M, int N, int K)
{
  __shared__ unsigned short As[2][128][40];   // stride 80B: conflict-free b128 reads
  __shared__ unsigned short Bs[2][64][40];

  const int tid  = threadIdx.x;
  const int m0   = blockIdx.y * 128;
  const int n0   = blockIdx.x * 64;
  const int lane = tid & 31, wave = tid >> 5;       // 8 waves, one 16-row strip each
  const int l15 = lane & 15, hl = lane >> 4;

  // async copy slot assignment: A = 512 b128 slots (2/thread), B = 256 (1/thread)
  const int arow0 = tid >> 2;                 // slot tid
  const int acol0 = (tid & 3) * 8;
  const int arow1 = (tid + 256) >> 2;         // slot tid+256
  const int acol1 = acol0;                    // (slot&3) identical
  const int brow  = tid >> 2;
  const int bcol  = (tid & 3) * 8;

  auto issue = [&](int k0, int buf) {
    async_ld_b128((unsigned)(size_t)&As[buf][arow0][acol0],
                  A + (size_t)(m0 + arow0) * K + k0 + acol0);
    async_ld_b128((unsigned)(size_t)&As[buf][arow1][acol1],
                  A + (size_t)(m0 + arow1) * K + k0 + acol1);
    async_ld_b128((unsigned)(size_t)&Bs[buf][brow][bcol],
                  Bt + (size_t)(n0 + brow) * K + k0 + bcol);
  };

  v8f acc[4] = {};
  const int nk = K >> 5;

  issue(0, 0);
  ASYNC_WAIT0();
  __syncthreads();

  for (int i = 0; i < nk; ++i) {
    const int cur = i & 1;
    if (i + 1 < nk) issue((i + 1) << 5, cur ^ 1);

    Frag16 a;
    a.h[0] = *(const v8us*)&As[cur][wave * 16 + l15][hl * 8];
    a.h[1] = *(const v8us*)&As[cur][wave * 16 + l15][16 + hl * 8];
    #pragma unroll
    for (int t = 0; t < 4; ++t) {
      Frag16 b;
      b.h[0] = *(const v8us*)&Bs[cur][t * 16 + l15][hl * 8];
      b.h[1] = *(const v8us*)&Bs[cur][t * 16 + l15][16 + hl * 8];
      acc[t] = wmma_bf16(a, b, acc[t]);
    }

    ASYNC_WAIT0();     // own async copies (next tile) done before the barrier
    __syncthreads();   // everyone's copies visible / current tile free
  }

  const int mbase = m0 + wave * 16 + 8 * hl;
  #pragma unroll
  for (int j = 0; j < 8; ++j) {
    const int m = mbase + j;
    #pragma unroll
    for (int t = 0; t < 4; ++t) {
      const int n = n0 + t * 16 + l15;
      const float v = acc[t][j];
      if (EPI == 0) {
        outB[(size_t)m * N + n] = f2bf(v + bias[n]);
      } else if (EPI == 1) {
        float x  = v + bias[n];
        int bb_  = m / S_;        int ss = m - bb_ * S_;
        int hd   = n / DH_;       int d  = n - hd * DH_;
        outB[(((size_t)bb_ * NH_ + hd) * DH_ + d) * S_ + ss] = f2bf(x);
      } else if (EPI == 2) {
        outF[(size_t)m * N + n] = v + bias[n] + resid[(size_t)m * N + n];
      } else {
        float x = v + bias[n];
        float gl = 0.5f * x * (1.0f + erff(x * 0.70710678118654752f));
        outB[(size_t)m * N + n] = f2bf(gl);
      }
    }
  }
}

// ---------------- fused flash-style attention ----------------
// one wave handles one (batch, head, 16-row Q tile); 4 waves / block
__global__ __launch_bounds__(128) void attn_kernel(
    const unsigned short* __restrict__ Q,    // [B*S][H] bf16 (head-interleaved)
    const unsigned short* __restrict__ Km,   // [B*S][H] bf16
    const unsigned short* __restrict__ Vt,   // [B][NH][DH][S] bf16
    const int* __restrict__ msk,             // [B][S]
    unsigned short* __restrict__ Ctx)        // [B*S][H] bf16
{
  __shared__ unsigned short Pl[4][16][40];   // per-wave P staging
  const int wave = threadIdx.x >> 5, lane = threadIdx.x & 31;
  const int l15 = lane & 15, hl = lane >> 4;
  const int gw = blockIdx.x * 4 + wave;
  const int qt = gw & 31;                    // S/16 = 32 q tiles
  const int hd = (gw >> 5) % NH_;
  const int b  = gw / (32 * NH_);
  const int q0 = qt * 16;

  const unsigned short* qb = Q  + ((size_t)b * S_) * H_ + hd * DH_;
  const unsigned short* kb = Km + ((size_t)b * S_) * H_ + hd * DH_;
  const unsigned short* vb = Vt + (((size_t)b * NH_ + hd) * DH_) * S_;

  // Q tile 16x64 kept in registers as two A-fragments (d 0..31, 32..63)
  Frag16 aq0, aq1;
  {
    const unsigned short* qr = qb + (size_t)(q0 + l15) * H_;
    aq0.h[0] = *(const v8us*)(qr + hl * 8);
    aq0.h[1] = *(const v8us*)(qr + 16 + hl * 8);
    aq1.h[0] = *(const v8us*)(qr + 32 + hl * 8);
    aq1.h[1] = *(const v8us*)(qr + 48 + hl * 8);
  }

  v8f o[4] = {};                 // ctx accumulators, 16x64
  float Mr[8], Lr[8];
  #pragma unroll
  for (int j = 0; j < 8; ++j) { Mr[j] = -1e30f; Lr[j] = 0.f; }
  const float scale = 0.125f;    // 1/sqrt(64)

  for (int kt = 0; kt < S_; kt += 32) {
    v8f s0 = {}, s1 = {};
    {  // keys kt..kt+15
      const unsigned short* kr = kb + (size_t)(kt + l15) * H_;
      Frag16 bA, bB;
      bA.h[0] = *(const v8us*)(kr + hl * 8);
      bA.h[1] = *(const v8us*)(kr + 16 + hl * 8);
      bB.h[0] = *(const v8us*)(kr + 32 + hl * 8);
      bB.h[1] = *(const v8us*)(kr + 48 + hl * 8);
      s0 = wmma_bf16(aq0, bA, s0);
      s0 = wmma_bf16(aq1, bB, s0);
    }
    {  // keys kt+16..kt+31
      const unsigned short* kr = kb + (size_t)(kt + 16 + l15) * H_;
      Frag16 bA, bB;
      bA.h[0] = *(const v8us*)(kr + hl * 8);
      bA.h[1] = *(const v8us*)(kr + 16 + hl * 8);
      bB.h[0] = *(const v8us*)(kr + 32 + hl * 8);
      bB.h[1] = *(const v8us*)(kr + 48 + hl * 8);
      s1 = wmma_bf16(aq0, bA, s1);
      s1 = wmma_bf16(aq1, bB, s1);
    }

    const float bias0 = (float)(1 - msk[b * S_ + kt + l15])      * -1e9f;
    const float bias1 = (float)(1 - msk[b * S_ + kt + 16 + l15]) * -1e9f;

    #pragma unroll
    for (int j = 0; j < 8; ++j) {            // row = 8*hl + j
      float v0 = s0[j] * scale + bias0;
      float v1 = s1[j] * scale + bias1;
      float rm = fmaxf(v0, v1);
      #pragma unroll
      for (int d = 8; d >= 1; d >>= 1) rm = fmaxf(rm, __shfl_xor(rm, d, 32));
      float nM   = fmaxf(Mr[j], rm);
      float corr = __expf(Mr[j] - nM);
      float p0 = __expf(v0 - nM);
      float p1 = __expf(v1 - nM);
      float rs = p0 + p1;
      #pragma unroll
      for (int d = 8; d >= 1; d >>= 1) rs += __shfl_xor(rs, d, 32);
      Lr[j] = Lr[j] * corr + rs;
      Mr[j] = nM;
      #pragma unroll
      for (int t = 0; t < 4; ++t) o[t][j] *= corr;
      Pl[wave][8 * hl + j][l15]      = f2bf(p0);
      Pl[wave][8 * hl + j][16 + l15] = f2bf(p1);
    }
    __asm__ volatile("s_wait_dscnt 0" ::: "memory");

    Frag16 ap;     // P 16x32 as A-fragment
    ap.h[0] = *(const v8us*)&Pl[wave][l15][hl * 8];
    ap.h[1] = *(const v8us*)&Pl[wave][l15][16 + hl * 8];

    #pragma unroll
    for (int t = 0; t < 4; ++t) {   // 4 d-subtiles of 16
      const unsigned short* vr = vb + (size_t)(t * 16 + l15) * S_ + kt + hl * 8;
      Frag16 bv_;
      bv_.h[0] = *(const v8us*)(vr);
      bv_.h[1] = *(const v8us*)(vr + 16);
      o[t] = wmma_bf16(ap, bv_, o[t]);
    }
  }

  #pragma unroll
  for (int j = 0; j < 8; ++j) {
    const float inv = 1.f / Lr[j];
    const int m = q0 + 8 * hl + j;
    unsigned short* crow = Ctx + ((size_t)b * S_ + m) * H_ + hd * DH_;
    #pragma unroll
    for (int t = 0; t < 4; ++t)
      crow[t * 16 + l15] = f2bf(o[t][j] * inv);
  }
}

// ---------------- stable valid-token compaction ----------------
__global__ void perm_kernel(const int* __restrict__ valid,
                            int* __restrict__ perm, int* __restrict__ counts)
{
  const int b = blockIdx.x;
  if (threadIdx.x == 0) {
    int c = 0;
    for (int s = 0; s < S_; ++s)
      if (valid[b * S_ + s]) perm[b * S_ + (c++)] = s;
    counts[b] = c;
  }
}

// ---------------- classifier head + softmax ----------------
__global__ __launch_bounds__(256) void cls_kernel(
    const float* __restrict__ h, const int* __restrict__ perm,
    const int* __restrict__ counts, const float* __restrict__ W,
    const float* __restrict__ bias, float* __restrict__ out)
{
  __shared__ float red8[8];
  __shared__ float logit[NL_];
  const int tok = blockIdx.x;
  const int b = tok >> 9;
  const int s = tok & (S_ - 1);
  const bool ok = s < counts[b];
  const float* row = h + ((size_t)b * S_ + (ok ? perm[b * S_ + s] : 0)) * H_;

  float part[NL_];
  #pragma unroll
  for (int c = 0; c < NL_; ++c) part[c] = 0.f;
  if (ok) {
    for (int i = threadIdx.x; i < H_; i += 256) {
      const float x = row[i];
      const float* wr = W + (size_t)i * NL_;
      #pragma unroll
      for (int c = 0; c < NL_; ++c) part[c] += x * wr[c];
    }
  }
  for (int c = 0; c < NL_; ++c) {
    float v = block_sum256(part[c], red8);
    if (threadIdx.x == 0) logit[c] = v + bias[c];
  }
  __syncthreads();
  if (threadIdx.x == 0) {
    float mx = -1e30f;
    for (int c = 0; c < NL_; ++c) mx = fmaxf(mx, logit[c]);
    float sm = 0.f, e[NL_];
    for (int c = 0; c < NL_; ++c) { e[c] = __expf(logit[c] - mx); sm += e[c]; }
    const float inv = 1.f / sm;
    for (int c = 0; c < NL_; ++c) out[(size_t)tok * NL_ + c] = e[c] * inv;
  }
}

// ---------------- host orchestration ----------------
extern "C" void kernel_launch(void* const* d_in, const int* in_sizes, int n_in,
                              void* d_out, int out_size, void* d_ws, size_t ws_size,
                              hipStream_t stream)
{
  (void)in_sizes; (void)n_in; (void)out_size; (void)ws_size;

  const int*   ids   = (const int*)d_in[0];
  const int*   msk   = (const int*)d_in[1];
  const int*   tids  = (const int*)d_in[2];
  const int*   vmask = (const int*)d_in[3];
  const float* wemb  = (const float*)d_in[4];
  const float* pemb  = (const float*)d_in[5];
  const float* temb  = (const float*)d_in[6];
  const float* eg    = (const float*)d_in[7];
  const float* ebta  = (const float*)d_in[8];
  const float* Wq    = (const float*)d_in[9];
  const float* bq    = (const float*)d_in[10];
  const float* Wk    = (const float*)d_in[11];
  const float* bk    = (const float*)d_in[12];
  const float* Wv    = (const float*)d_in[13];
  const float* bv    = (const float*)d_in[14];
  const float* Wo    = (const float*)d_in[15];
  const float* bo    = (const float*)d_in[16];
  const float* ln1g  = (const float*)d_in[17];
  const float* ln1b  = (const float*)d_in[18];
  const float* W1    = (const float*)d_in[19];
  const float* W1b   = (const float*)d_in[20];
  const float* W2    = (const float*)d_in[21];
  const float* W2b   = (const float*)d_in[22];
  const float* ln2g  = (const float*)d_in[23];
  const float* ln2b  = (const float*)d_in[24];
  const float* clsW  = (const float*)d_in[25];
  const float* clsb  = (const float*)d_in[26];

  // workspace carve-out
  char* ws = (char*)d_ws;
  size_t off = 0;
  auto carve = [&](size_t bytes) -> char* {
    char* p = ws + off;
    off += (bytes + 255) & ~(size_t)255;
    return p;
  };
  float*          h    = (float*)carve((size_t)M_ * H_ * 4);
  float*          tmp  = (float*)carve((size_t)M_ * H_ * 4);
  unsigned short* xbf  = (unsigned short*)carve((size_t)M_ * H_ * 2);
  unsigned short* qbf  = (unsigned short*)carve((size_t)M_ * H_ * 2);
  unsigned short* kbf  = (unsigned short*)carve((size_t)M_ * H_ * 2);
  unsigned short* vtb  = (unsigned short*)carve((size_t)M_ * H_ * 2);
  unsigned short* ctx  = (unsigned short*)carve((size_t)M_ * H_ * 2);
  unsigned short* ffb  = (unsigned short*)carve((size_t)M_ * FF_ * 2);
  unsigned short* wst  = (unsigned short*)carve((size_t)FF_ * H_ * 2);
  int*            perm = (int*)carve((size_t)B_ * S_ * 4);
  int*            cnts = (int*)carve(64);

  const dim3 blk256(256), blk128(128);
  const dim3 gemmHH(H_ / 64, M_ / 128);   // (12,64)
  const dim3 gemmHF(FF_ / 64, M_ / 128);  // (48,64)
  const dim3 cvtHH(H_ / 32, H_ / 32);     // W[H][H]
  const dim3 cvtHF(FF_ / 32, H_ / 32);    // W1[H][FF]
  const dim3 cvtFH(H_ / 32, FF_ / 32);    // W2[FF][H]
  const int attnBlocks = (B_ * NH_ * (S_ / 16)) / 4;  // 1536

  embed_ln_kernel<<<M_, blk256, 0, stream>>>(ids, tids, wemb, pemb, temb,
                                             eg, ebta, h, xbf);

  for (int l = 0; l < L_; ++l) {
    const float* wq = Wq + (size_t)l * H_ * H_;
    const float* wk = Wk + (size_t)l * H_ * H_;
    const float* wv = Wv + (size_t)l * H_ * H_;
    const float* wo = Wo + (size_t)l * H_ * H_;
    const float* w1 = W1 + (size_t)l * H_ * FF_;
    const float* w2 = W2 + (size_t)l * FF_ * H_;

    convert_wt_kernel<<<cvtHH, blk256, 0, stream>>>(wq, wst, H_, H_);
    gemm_nt_bf16<0><<<gemmHH, blk256, 0, stream>>>(xbf, wst, bq + l * H_,
                                                   nullptr, nullptr, qbf, M_, H_, H_);
    convert_wt_kernel<<<cvtHH, blk256, 0, stream>>>(wk, wst, H_, H_);
    gemm_nt_bf16<0><<<gemmHH, blk256, 0, stream>>>(xbf, wst, bk + l * H_,
                                                   nullptr, nullptr, kbf, M_, H_, H_);
    convert_wt_kernel<<<cvtHH, blk256, 0, stream>>>(wv, wst, H_, H_);
    gemm_nt_bf16<1><<<gemmHH, blk256, 0, stream>>>(xbf, wst, bv + l * H_,
                                                   nullptr, nullptr, vtb, M_, H_, H_);

    attn_kernel<<<attnBlocks, blk128, 0, stream>>>(qbf, kbf, vtb, msk, ctx);

    convert_wt_kernel<<<cvtHH, blk256, 0, stream>>>(wo, wst, H_, H_);
    gemm_nt_bf16<2><<<gemmHH, blk256, 0, stream>>>(ctx, wst, bo + l * H_,
                                                   h, tmp, nullptr, M_, H_, H_);
    ln_kernel<<<M_, blk256, 0, stream>>>(tmp, ln1g + l * H_, ln1b + l * H_, h, xbf);

    convert_wt_kernel<<<cvtHF, blk256, 0, stream>>>(w1, wst, H_, FF_);
    gemm_nt_bf16<3><<<gemmHF, blk256, 0, stream>>>(xbf, wst, W1b + l * FF_,
                                                   nullptr, nullptr, ffb, M_, FF_, H_);
    convert_wt_kernel<<<cvtFH, blk256, 0, stream>>>(w2, wst, FF_, H_);
    gemm_nt_bf16<2><<<gemmHH, blk256, 0, stream>>>(ffb, wst, W2b + l * H_,
                                                   h, tmp, nullptr, M_, H_, FF_);
    ln_kernel<<<M_, blk256, 0, stream>>>(tmp, ln2g + l * H_, ln2b + l * H_, h, xbf);
  }

  perm_kernel<<<B_, 32, 0, stream>>>(vmask, perm, cnts);
  cls_kernel<<<M_, blk256, 0, stream>>>(h, perm, cnts, clsW, clsb, (float*)d_out);
}
